// GCN_83373905150072
// MI455X (gfx1250) — matrix operand order repaired and checked by reading, not compiled, not run on hardware
//
#include <hip/hip_runtime.h>

// ---------------------------------------------------------------------------
// Dense-GCN stack on MI455X (gfx1250).
// Memory-bound on streaming A [32,1024,1024] eight times. Strategy:
//  - normalize A once, store as f16 (67 MB) -> fits in 192 MB L2, halves bytes
//  - all 8 A-propagations as v_wmma_f32_16x16x32_f16 with fp32 accumulate
//  - BN stats fused into the WMMA epilogue (shfl_xor reduce + f32 atomics)
// ---------------------------------------------------------------------------

typedef _Float16 v16h  __attribute__((ext_vector_type(16)));
typedef float    v8f   __attribute__((ext_vector_type(8)));
typedef _Float16 half8 __attribute__((ext_vector_type(8)));
typedef _Float16 half4 __attribute__((ext_vector_type(4)));
typedef float    f32x4 __attribute__((ext_vector_type(4)));

#define NB    32
#define NN    1024
#define EPSBN 1e-5f

// --- degree: dinv[b,n] = rsqrt(max(rowsum(A with diag=1), 1)) ----------------
__global__ void __launch_bounds__(256)
k_degree(const float* __restrict__ adj, float* __restrict__ dinv) {
    int wave = threadIdx.x >> 5;
    int lane = threadIdx.x & 31;
    int gr   = blockIdx.x * 8 + wave;            // 0..32767  (b*1024 + row)
    int b    = gr >> 10, row = gr & 1023;
    const float* arow = adj + ((size_t)b * NN + row) * NN;
    float s = 0.f;
    for (int j = lane; j < NN; j += 32)
        s += (j == row) ? 1.0f : arow[j];
    for (int m = 16; m; m >>= 1) s += __shfl_xor(s, m, 32);
    if (lane == 0) dinv[gr] = rsqrtf(fmaxf(s, 1.0f));
}

// --- A16[b,i,j] = f16( dinv_i * adj_ij(diag=1) * dinv_j ) --------------------
__global__ void __launch_bounds__(256)
k_scaleA(const float* __restrict__ adj, const float* __restrict__ dinv,
         _Float16* __restrict__ A16) {
    size_t t  = (size_t)blockIdx.x * blockDim.x + threadIdx.x;  // B*N*256 threads
    int   j4  = (int)(t & 255) * 4;
    int   row = (int)(t >> 8) & 1023;
    int   b   = (int)(t >> 18);
    size_t base = ((size_t)b * NN + row) * NN + j4;
    f32x4 a4 = *(const f32x4*)(adj + base);
    float di = dinv[b * NN + row];
    half4 hv;
    for (int u = 0; u < 4; ++u) {
        int j = j4 + u;
        float v = (j == row) ? 1.0f : a4[u];
        hv[u] = (_Float16)(di * v * dinv[b * NN + j]);
    }
    *(half4*)(A16 + base) = hv;
}

__global__ void k_zero(float* __restrict__ p, int n) {
    int i = blockIdx.x * blockDim.x + threadIdx.x;
    if (i < n) p[i] = 0.f;
}

// --- fused (optional BN of hin) + channel-mix hin@W -> f16 [B,N,16] padded ---
// Also optionally writes the normalized input into hout (concat buffer, ld 6).
__global__ void __launch_bounds__(256)
k_xw(const float* __restrict__ hin, int ldin, int Cin,
     const float* __restrict__ W, const float* __restrict__ stats, float invCnt,
     float* __restrict__ hout, int houtOff, int Cout,
     _Float16* __restrict__ xw16) {
    int t = blockIdx.x * blockDim.x + threadIdx.x;   // over B*N = 32768
    if (t >= NB * NN) return;
    int m = t & 1023;
    float mean = 0.f, scale = 1.f;
    if (stats) {
        float s1 = stats[m], s2 = stats[NN + m];
        mean = s1 * invCnt;
        float var = s2 * invCnt - mean * mean;
        scale = rsqrtf(var + EPSBN);
    }
    const float* hp = hin + (size_t)t * ldin;
    float in[32];
    for (int i = 0; i < Cin; ++i) {
        float v = (hp[i] - mean) * scale;
        in[i] = v;
        if (hout) hout[(size_t)t * 6 + houtOff + i] = v;
    }
    union { _Float16 a[16]; half8 h[2]; } o;
    for (int c = 0; c < 16; ++c) {
        float acc = 0.f;
        if (c < Cout)
            for (int i = 0; i < Cin; ++i) acc += in[i] * W[i * Cout + c];
        o.a[c] = (_Float16)acc;
    }
    half8* dst = (half8*)(xw16 + (size_t)t * 16);
    dst[0] = o.h[0];
    dst[1] = o.h[1];
}

// --- Y = relu(A16[b] @ XW16[b] + bias); accumulate per-node BN sum/sumsq -----
// One wave per 16-row M-tile; K loop of 32 x v_wmma_f32_16x16x32_f16.
__global__ void __launch_bounds__(256)
k_gcn_wmma(const _Float16* __restrict__ A16, const _Float16* __restrict__ XW16,
           const float* __restrict__ bias, int Cout,
           float* __restrict__ Y, float* __restrict__ stats) {
    int wid   = (blockIdx.x << 3) + (threadIdx.x >> 5);   // 0..2047 tiles
    int lane  = threadIdx.x & 31;
    int b     = wid >> 6;
    int tileM = wid & 63;

    const _Float16* Ab = A16 + ((size_t)b << 20);          // 1024*1024 per batch
    const _Float16* Xb = XW16 + ((size_t)b << 14);         // 1024*16 per batch

    // A fragment (16x32 f16): lane row = tileM*16 + (lane&15);
    // lanes 0-15 take K 0..7 / 16..23, lanes 16-31 take K 8..15 / 24..31.
    int arow = (tileM << 4) + (lane & 15);
    const _Float16* Aro = Ab + ((size_t)arow << 10) + ((lane >> 4) << 3);
    // B fragment (32x16 f16): lane col n = lane&15, same K order as A.
    const _Float16* Xro = Xb + (lane & 15) + ((size_t)((lane >> 4) << 3) << 4);

    v8f acc = {0.f, 0.f, 0.f, 0.f, 0.f, 0.f, 0.f, 0.f};
    union U { v16h v; half8 h[2]; };
    for (int k0 = 0; k0 < NN; k0 += 32) {
        U a, x;
        a.h[0] = *(const half8*)(Aro + k0);
        a.h[1] = *(const half8*)(Aro + k0 + 16);
        const _Float16* xr = Xro + (size_t)k0 * 16;
        for (int s = 0; s < 8; ++s) {
            x.v[s] = xr[s * 16];
            x.v[s + 8] = xr[(s + 16) * 16];
        }
        acc = __builtin_amdgcn_wmma_f32_16x16x32_f16(
            false, a.v, false, x.v, (short)0, acc, false, false);
    }

    // C/D layout: VGPR r -> row tileM*16 + r + (lane<16?0:8), col = lane&15
    int n     = lane & 15;
    int mbase = (tileM << 4) + ((lane >> 4) << 3);
    float bn  = (n < Cout) ? bias[n] : 0.f;
    for (int r = 0; r < 8; ++r) {
        int   m  = mbase + r;
        float v  = fmaxf(acc[r] + bn, 0.f);
        float vm = (n < Cout) ? v : 0.f;
        if (n < Cout) Y[((size_t)b * NN + m) * Cout + n] = v;
        float s = vm, q = vm * vm;
        for (int msk = 1; msk < 16; msk <<= 1) {
            s += __shfl_xor(s, msk, 32);
            q += __shfl_xor(q, msk, 32);
        }
        if (n == 0) {
            atomicAdd(&stats[m], s);
            atomicAdd(&stats[NN + m], q);
        }
    }
}

// --- max over nodes: pooled[b*6+c] = max_n h[b,n,c] --------------------------
__global__ void __launch_bounds__(256)
k_pool(const float* __restrict__ h, float* __restrict__ pooled) {
    int b = blockIdx.x / 6, c = blockIdx.x % 6;
    const float* p = h + (size_t)b * NN * 6 + c;
    float mx = -3.402823466e38f;
    for (int n = threadIdx.x; n < NN; n += blockDim.x)
        mx = fmaxf(mx, p[(size_t)n * 6]);
    __shared__ float sm[256];
    sm[threadIdx.x] = mx;
    __syncthreads();
    for (int s = 128; s; s >>= 1) {
        if (threadIdx.x < s) sm[threadIdx.x] = fmaxf(sm[threadIdx.x], sm[threadIdx.x + s]);
        __syncthreads();
    }
    if (threadIdx.x == 0) pooled[b * 6 + c] = sm[0];
}

// --- out[b,j] = bl[j] + sum_k pooled_concat[b,k] * Wl[k,j] -------------------
__global__ void k_final(const float* __restrict__ pooled, const float* __restrict__ Wl,
                        const float* __restrict__ bl, float* __restrict__ out) {
    int t = threadIdx.x;
    if (t >= NB * 3) return;
    int b = t / 3, j = t % 3;
    float s = bl[j];
    for (int blk = 0; blk < 4; ++blk)
        for (int c = 0; c < 6; ++c)
            s += pooled[blk * (NB * 6) + b * 6 + c] * Wl[(blk * 6 + c) * 3 + j];
    out[t] = s;
}

extern "C" void kernel_launch(void* const* d_in, const int* in_sizes, int n_in,
                              void* d_out, int out_size, void* d_ws, size_t ws_size,
                              hipStream_t stream) {
    const float* x   = (const float*)d_in[0];
    const float* adj = (const float*)d_in[1];
    const float* W1  = (const float*)d_in[2];
    const float* b1  = (const float*)d_in[3];
    const float* W2  = (const float*)d_in[4];
    const float* b2  = (const float*)d_in[5];
    const float* W3  = (const float*)d_in[6];
    const float* b3  = (const float*)d_in[7];
    const float* W5  = (const float*)d_in[8];
    const float* b5  = (const float*)d_in[9];
    const float* W7  = (const float*)d_in[10];
    const float* b7  = (const float*)d_in[11];
    const float* Wl  = (const float*)d_in[12];
    const float* bl  = (const float*)d_in[13];
    float* out = (float*)d_out;
    (void)in_sizes; (void)n_in; (void)out_size; (void)ws_size;

    // Workspace layout (bytes):
    char* ws = (char*)d_ws;
    _Float16* A16   = (_Float16*)(ws);                 // 32*1024*1024*2 = 67,108,864
    _Float16* XW16  = (_Float16*)(ws + 67108864);      // 32*1024*16*2   =  1,048,576
    float*    Y     = (float*)   (ws + 68157440);      // 32*1024*4*4    =    524,288
    float*    hA    = (float*)   (ws + 68681728);      // 32*1024*6*4    =    786,432
    float*    hB    = (float*)   (ws + 69468160);      // 32*1024*6*4    =    786,432
    float*    dinv  = (float*)   (ws + 70254592);      // 32*1024*4      =    131,072
    float*    stats = (float*)   (ws + 70385664);      // 8 convs * 2048 floats = 65,536
    float*    pooled= (float*)   (ws + 70451200);      // 4 * 192 floats =      3,072

    // 1) normalization data + f16 A (L2-resident for all 8 propagations)
    k_degree<<<4096, 256, 0, stream>>>(adj, dinv);
    k_scaleA<<<32768, 256, 0, stream>>>(adj, dinv, A16);
    k_zero  <<<68, 256, 0, stream>>>(stats, 8 * 2048 + 4 * 192);  // stats+pooled contiguous

    const float* Wa[4] = {W1, W3, W5, W7};
    const float* ba[4] = {b1, b3, b5, b7};
    float* hbufs[2] = {hA, hB};

    const float* hcur = x;
    int ldcur = 32, Cin = 32;

    for (int blk = 0; blk < 4; ++blk) {
        float* hnext = hbufs[blk & 1];
        float* st1 = stats + blk * 4096;
        float* st2 = stats + blk * 4096 + 2048;

        // conv1: XW = hcur @ Wa  -> f16 padded; Y1 = relu(A@XW + b); stats1
        k_xw<<<128, 256, 0, stream>>>(hcur, ldcur, Cin, Wa[blk],
                                      nullptr, 0.f, nullptr, 0, 4, XW16);
        k_gcn_wmma<<<256, 256, 0, stream>>>(A16, XW16, ba[blk], 4, Y, st1);

        // BN(Y1) -> x1 (hnext[:,0:4]); XW = x1 @ W2 ; Y2 = relu(A@XW + b2); stats2
        k_xw<<<128, 256, 0, stream>>>(Y, 4, 4, W2,
                                      st1, 1.0f / (NB * 4), hnext, 0, 2, XW16);
        k_gcn_wmma<<<256, 256, 0, stream>>>(A16, XW16, b2, 2, Y, st2);

        // BN(Y2) -> x2 (hnext[:,4:6]); no further mix needed (Cout=0)
        k_xw<<<128, 256, 0, stream>>>(Y, 2, 2, W2,
                                      st2, 1.0f / (NB * 2), hnext, 4, 0, XW16);

        // max-pool over nodes for this block's concat output
        k_pool<<<192, 256, 0, stream>>>(hnext, pooled + blk * (NB * 6));

        hcur = hnext; ldcur = 6; Cin = 6;
    }

    k_final<<<1, 128, 0, stream>>>(pooled, Wl, bl, out);
}
